// QueryCluster_30889404793011
// MI455X (gfx1250) — matrix operand order repaired and checked by reading, not compiled, not run on hardware
//
#include <hip/hip_runtime.h>
#include <hip/hip_bf16.h>
#include <math.h>

// ---------------- problem constants (from reference) ----------------
#define N_WAY      64
#define K_SHOT     5
#define Q_QUERY    128
#define D_FEAT     128
#define M_Q        (N_WAY * Q_QUERY)        // 8192 query rows
#define ROW_STRIDE (K_SHOT + Q_QUERY)       // 133 rows per class in x
#define K_TOPK     9
#define EPSF       1e-8f

typedef __attribute__((ext_vector_type(16))) _Float16 v16h;
typedef __attribute__((ext_vector_type(8)))  float    v8f;
typedef __attribute__((ext_vector_type(2)))  _Float16 h2;

// ---------------------------------------------------------------
// Fragment loaders for v_wmma_f32_16x16x32_f16 (wave32; this code's own
// contiguous K order, used for A and B alike). Rows are row-major, K contiguous, so
// each (2v,2v+1) K-pair is one aligned 32-bit load (compiler merges
// the contiguous pairs into global_load_b128s).
//
// A (16x32, MxK): lane L -> row M = L%16. VGPR v holds K-pair:
//   K = k0 + (L>=16 ? 16 : 0) + 2*v
// B (32x16, KxN): lane L -> col N = L%16 (a qn row, since B = qn^T).
//   VGPR v holds K-pair: K = k0 + (L>=16 ? 16 : 0) + 2*v
// ---------------------------------------------------------------
__device__ __forceinline__ v16h load_a_frag(const _Float16* __restrict__ row,
                                            int k0, int hi8) {
  v16h a;
#pragma unroll
  for (int v = 0; v < 8; ++v) {
    const int kk = k0 + (hi8 << 1) + (v << 1);
    h2 p = *(const h2*)(row + kk);
    a[2 * v] = p[0];
    a[2 * v + 1] = p[1];
  }
  return a;
}

__device__ __forceinline__ v16h load_b_frag(const _Float16* __restrict__ col,
                                            int k0, int hi16) {
  v16h b;
#pragma unroll
  for (int v = 0; v < 8; ++v) {
    const int kk = k0 + hi16 + (v << 1);
    h2 p = *(const h2*)(col + kk);
    b[2 * v] = p[0];
    b[2 * v + 1] = p[1];
  }
  return b;
}

// ---------------------------------------------------------------
// K1: prototypes = mean of 5 support rows, l2-normalized, stored f16.
// ---------------------------------------------------------------
__global__ __launch_bounds__(128) void proto_kernel(const float* __restrict__ x,
                                                    _Float16* __restrict__ protoN16) {
  const int p = blockIdx.x;   // class 0..63
  const int t = threadIdx.x;  // feature dim
  float s = 0.f;
#pragma unroll
  for (int sh = 0; sh < K_SHOT; ++sh)
    s += x[((size_t)(p * ROW_STRIDE + sh)) * D_FEAT + t];
  s *= (1.0f / K_SHOT);

  __shared__ float red[128];
  red[t] = s * s;
  __syncthreads();
  for (int off = 64; off > 0; off >>= 1) {
    if (t < off) red[t] += red[t + off];
    __syncthreads();
  }
  const float nrm = fmaxf(sqrtf(red[0]), EPSF);
  protoN16[p * D_FEAT + t] = (_Float16)(s / nrm);
}

// ---------------------------------------------------------------
// K2: qn = l2norm(query) stored f16 (WMMA operand), 1 block per row.
// query row i -> x row (i/128)*133 + 5 + (i%128)
// ---------------------------------------------------------------
__global__ __launch_bounds__(128) void qnorm_kernel(const float* __restrict__ x,
                                                    _Float16* __restrict__ qn16) {
  const int i = blockIdx.x;
  const int t = threadIdx.x;
  const int cls = i >> 7;
  const int qq  = i & 127;
  const float v = x[((size_t)(cls * ROW_STRIDE + K_SHOT + qq)) * D_FEAT + t];

  __shared__ float red[128];
  red[t] = v * v;
  __syncthreads();
  for (int off = 64; off > 0; off >>= 1) {
    if (t < off) red[t] += red[t + off];
    __syncthreads();
  }
  const float nrm = fmaxf(sqrtf(red[0]), EPSF);
  qn16[(size_t)i * D_FEAT + t] = (_Float16)(v / nrm);
}

// ---------------------------------------------------------------
// K3: fused similarity GEMM + per-row top-9 (one wave / 16 rows).
//  * A fragments (4 x v16h) live in registers for the whole sweep.
//  * Software-pipelined: the NEXT column tile's 8 B-loads are issued
//    before the current tile's 4 WMMAs + scan, hiding load latency
//    under matrix ops (B stream is L2-resident: 2 MB).
//  * Top-9 kept sorted in registers; candidates pre-filtered against
//    the current 9th-best so the 9-deep insertion network is emitted
//    once (scan loop NOT unrolled) and rarely executed.
// ---------------------------------------------------------------
__global__ __launch_bounds__(32) void sim_topk_kernel(const _Float16* __restrict__ qn16,
                                                      float* __restrict__ tkVal,
                                                      int* __restrict__ tkIdx) {
  __shared__ float tile[16][17];  // +1 pad: dodge bank conflicts
  const int rb   = blockIdx.x;    // row block 0..511
  const int lane = threadIdx.x;
  const int m    = lane & 15;
  const int hi   = lane >> 4;
  const int hi8  = hi ? 8 : 0;
  const int hi16 = hi ? 16 : 0;

  float tv[K_TOPK];
  int   ti[K_TOPK];
#pragma unroll
  for (int s = 0; s < K_TOPK; ++s) { tv[s] = -3.0e38f; ti[s] = 0; }

  const _Float16* Arow = qn16 + (size_t)(rb * 16 + m) * D_FEAT;
  const v16h a0 = load_a_frag(Arow,  0, hi8);
  const v16h a1 = load_a_frag(Arow, 32, hi8);
  const v16h a2 = load_a_frag(Arow, 64, hi8);
  const v16h a3 = load_a_frag(Arow, 96, hi8);

  // prologue: B fragments for column tile 0
  const _Float16* B0 = qn16 + (size_t)m * D_FEAT;
  v16h b0 = load_b_frag(B0,  0, hi16);
  v16h b1 = load_b_frag(B0, 32, hi16);
  v16h b2 = load_b_frag(B0, 64, hi16);
  v16h b3 = load_b_frag(B0, 96, hi16);

#pragma unroll 1
  for (int cb = 0; cb < M_Q / 16; ++cb) {
    // issue next tile's B loads now; they retire under the WMMAs + scan
    const int nc = (cb + 1 < M_Q / 16) ? (cb + 1) : 0;  // wrap: harmless
    const _Float16* Bn = qn16 + (size_t)(nc * 16 + m) * D_FEAT;
    v16h nb0 = load_b_frag(Bn,  0, hi16);
    v16h nb1 = load_b_frag(Bn, 32, hi16);
    v16h nb2 = load_b_frag(Bn, 64, hi16);
    v16h nb3 = load_b_frag(Bn, 96, hi16);

    v8f acc = {};
    acc = __builtin_amdgcn_wmma_f32_16x16x32_f16(false, a0, false, b0, (short)0, acc, false, false);
    acc = __builtin_amdgcn_wmma_f32_16x16x32_f16(false, a1, false, b1, (short)0, acc, false, false);
    acc = __builtin_amdgcn_wmma_f32_16x16x32_f16(false, a2, false, b2, (short)0, acc, false, false);
    acc = __builtin_amdgcn_wmma_f32_16x16x32_f16(false, a3, false, b3, (short)0, acc, false, false);

    // D layout: lane L, VGPR v -> row M = v + 8*(L>=16), col N = L%16
#pragma unroll
    for (int v = 0; v < 8; ++v) tile[v + hi8][m] = acc[v];
    __syncthreads();

    if (lane < 16) {
      float thresh = tv[K_TOPK - 1];
#pragma unroll 1
      for (int c = 0; c < 16; ++c) {
        float cv = tile[lane][c];
        if (cv > thresh) {
          int ci = cb * 16 + c;
#pragma unroll
          for (int s = 0; s < K_TOPK; ++s) {
            if (cv > tv[s]) {
              const float t1 = tv[s]; const int t2 = ti[s];
              tv[s] = cv; ti[s] = ci;
              cv = t1; ci = t2;
            }
          }
          thresh = tv[K_TOPK - 1];
        }
      }
    }
    __syncthreads();

    b0 = nb0; b1 = nb1; b2 = nb2; b3 = nb3;
  }

  if (lane < 16) {
    const int r = rb * 16 + lane;
#pragma unroll
    for (int s = 0; s < K_TOPK; ++s) {
      tkVal[r * K_TOPK + s] = tv[s];
      tkIdx[r * K_TOPK + s] = ti[s];
    }
  }
}

// ---------------------------------------------------------------
// K4: mutual-kNN mask + softmax + sparse (weight @ query) + l2norm.
// weight row i has <=9 nonzeros (diagonal always mutual), so this is a
// tiny gather: one block (128 threads = one dim each) per query row.
// Result stored f16 for the final WMMA GEMM.
// ---------------------------------------------------------------
__global__ __launch_bounds__(128) void adapt_kernel(const float* __restrict__ x,
                                                    const float* __restrict__ tkVal,
                                                    const int* __restrict__ tkIdx,
                                                    _Float16* __restrict__ aq16) {
  const int i = blockIdx.x;   // query row
  const int t = threadIdx.x;  // feature dim

  __shared__ float w[K_TOPK];
  __shared__ int   jidx[K_TOPK];
  __shared__ int   valid[K_TOPK];
  __shared__ float red[128];

  if (t < K_TOPK) {
    const int j = tkIdx[i * K_TOPK + t];
    jidx[t] = j;
    int f = 0;
#pragma unroll
    for (int s = 0; s < K_TOPK; ++s)
      if (tkIdx[j * K_TOPK + s] == i) f = 1;
    valid[t] = f;
    w[t] = tkVal[i * K_TOPK + t];
  }
  __syncthreads();

  if (t == 0) {
    float mx = -3.0e38f;
#pragma unroll
    for (int s = 0; s < K_TOPK; ++s)
      if (valid[s] && w[s] > mx) mx = w[s];
    float e[K_TOPK];
    float sum = 0.f;
#pragma unroll
    for (int s = 0; s < K_TOPK; ++s) {
      e[s] = valid[s] ? __expf(w[s] - mx) : 0.f;
      sum += e[s];
    }
    const float inv = 1.f / fmaxf(sum, 1e-30f);
#pragma unroll
    for (int s = 0; s < K_TOPK; ++s) w[s] = e[s] * inv;
  }
  __syncthreads();

  float acc = 0.f;
#pragma unroll
  for (int s = 0; s < K_TOPK; ++s) {
    const float ws = w[s];
    if (ws != 0.f) {
      const int j   = jidx[s];
      const int cls = j >> 7;
      const int qq  = j & 127;
      acc += ws * x[((size_t)(cls * ROW_STRIDE + K_SHOT + qq)) * D_FEAT + t];
    }
  }

  red[t] = acc * acc;
  __syncthreads();
  for (int off = 64; off > 0; off >>= 1) {
    if (t < off) red[t] += red[t + off];
    __syncthreads();
  }
  const float nrm = fmaxf(sqrtf(red[0]), EPSF);
  aq16[(size_t)i * D_FEAT + t] = (_Float16)(acc / nrm);
}

// ---------------------------------------------------------------
// K5: out = tao * l2norm(adapted) @ protoN^T  (8192x64x128 GEMM),
// WMMA tiles of 16x16, K=128 -> 4 wmma ops per tile, tao read on device.
// ---------------------------------------------------------------
__global__ __launch_bounds__(32) void out_gemm_kernel(const _Float16* __restrict__ aq16,
                                                      const _Float16* __restrict__ protoN16,
                                                      const float* __restrict__ taoPtr,
                                                      float* __restrict__ out) {
  const int rb   = blockIdx.x;   // 0..511 row tiles
  const int cb   = blockIdx.y;   // 0..3 col tiles
  const int lane = threadIdx.x;
  const int m    = lane & 15;
  const int hi   = lane >> 4;
  const int hi8  = hi ? 8 : 0;
  const int hi16 = hi ? 16 : 0;
  const float tao = *taoPtr;

  const _Float16* Arow = aq16     + (size_t)(rb * 16 + m) * D_FEAT;
  const _Float16* Brow = protoN16 + (size_t)(cb * 16 + m) * D_FEAT;

  v8f acc = {};
#pragma unroll
  for (int k0 = 0; k0 < D_FEAT; k0 += 32) {
    v16h a = load_a_frag(Arow, k0, hi8);
    v16h b = load_b_frag(Brow, k0, hi16);
    acc = __builtin_amdgcn_wmma_f32_16x16x32_f16(false, a, false, b,
                                                 (short)0, acc, false, false);
  }
#pragma unroll
  for (int v = 0; v < 8; ++v) {
    const int r = rb * 16 + v + hi8;
    const int c = cb * 16 + m;
    out[(size_t)r * N_WAY + c] = tao * acc[v];
  }
}

// ---------------------------------------------------------------
// Workspace layout (bytes from d_ws base), total ~4.6 MB:
//   [0,        2097152) qn16       8192x128 f16
//   [2097152,  2113536) protoN16     64x128 f16
//   [2113536,  2408448) tkVal      8192x9   f32
//   [2408448,  2703360) tkIdx      8192x9   i32
//   [2703360,  4800512) aq16       8192x128 f16
// ---------------------------------------------------------------
extern "C" void kernel_launch(void* const* d_in, const int* in_sizes, int n_in,
                              void* d_out, int out_size, void* d_ws, size_t ws_size,
                              hipStream_t stream) {
  const float* x   = (const float*)d_in[0];
  const float* tao = (const float*)d_in[1];
  (void)in_sizes; (void)n_in; (void)out_size; (void)ws_size;

  char* ws = (char*)d_ws;
  _Float16* qn16     = (_Float16*)(ws + 0);
  _Float16* protoN16 = (_Float16*)(ws + 2097152);
  float*    tkVal    = (float*)   (ws + 2113536);
  int*      tkIdx    = (int*)     (ws + 2408448);
  _Float16* aq16     = (_Float16*)(ws + 2703360);
  float*    out      = (float*)d_out;

  proto_kernel<<<N_WAY, 128, 0, stream>>>(x, protoN16);
  qnorm_kernel<<<M_Q, 128, 0, stream>>>(x, qn16);
  sim_topk_kernel<<<M_Q / 16, 32, 0, stream>>>(qn16, tkVal, tkIdx);
  adapt_kernel<<<M_Q, 128, 0, stream>>>(x, tkVal, tkIdx, aq16);
  out_gemm_kernel<<<dim3(M_Q / 16, N_WAY / 16), 32, 0, stream>>>(aq16, protoN16, tao, out);
}
